// LinearMLPNode_79379585565298
// MI455X (gfx1250) — matrix operand order repaired and checked by reading, not compiled, not run on hardware
//
#include <hip/hip_runtime.h>
#include <cstdint>
#include <cstddef>

// ---- problem constants (from reference) ----
#define NODES   64
#define KDIM    256     // F
#define ODIM    256     // O
#define BATCH   4096    // graphs

// ---- tiling ----
#define BM 128          // batch rows per block
#define BN 256          // output cols per block (all of O -> x streamed once)
#define BK 32           // K chunk (f16 WMMA K)
#define KP (BK + 8)     // LDS pitch in f16 elems (+16B pad: conflict-free, 16B-aligned rows)
#define WM 32           // wave M tile (2 frags)
#define WN 128          // wave N tile (8 frags)

typedef __attribute__((ext_vector_type(16))) _Float16 v16h;
typedef __attribute__((ext_vector_type(8)))  _Float16 v8h;
typedef __attribute__((ext_vector_type(4)))  _Float16 v4h;
typedef __attribute__((ext_vector_type(8)))  float    v8f;
typedef __attribute__((ext_vector_type(4)))  float    v4f;

static __device__ __forceinline__ v16h cat8(v8h a, v8h b) {
  return __builtin_shufflevector(a, b, 0,1,2,3,4,5,6,7,8,9,10,11,12,13,14,15);
}

// split fp32 into f16 hi (RNE) + f16 lo (residual): ~22 effective mantissa bits
static __device__ __forceinline__ void split2(float x, _Float16& hi, _Float16& lo) {
  hi = (_Float16)x;
  lo = (_Float16)(x - (float)hi);
}

__global__ __launch_bounds__(256, 1)
void node_linear_f16x3(const float* __restrict__ x,
                       const float* __restrict__ W,
                       float* __restrict__ out)
{
  // A: row-major [m][k]; B: column-major [n][k] so WMMA frags are plain ds_load_b128
  __shared__ _Float16 Ahi[2][BM][KP];   // 20 KB
  __shared__ _Float16 Alo[2][BM][KP];   // 20 KB
  __shared__ _Float16 Bhi[2][BN][KP];   // 40 KB
  __shared__ _Float16 Blo[2][BN][KP];   // 40 KB

  const int t    = threadIdx.x;
  const int lane = t & 31;
  const int wid  = t >> 5;
  const int node = blockIdx.y;
  const int bm0  = blockIdx.x * BM;       // batch-row block

  const int wm = (wid & 3) * WM;          // wave M offset: 0,32,64,96
  const int wn = (wid >> 2) * WN;         // wave N offset: 0,128
  const int ml = lane & 15;
  const int kh = lane >> 4;               // K-half select per ISA frag layouts

  v8f acc[2][8] = {};

  auto stage = [&](int kt, int buf) {
    const int k0 = kt * BK;
    // ---- stage A: x[(bm0+m)*64 + node][k0 .. k0+31], 16 floats/thread ----
    #pragma unroll
    for (int p = 0; p < 4; ++p) {
      const int m  = p * 32 + (t >> 3);      // 0..127
      const int kk = (t & 7) * 4;            // 0..28
      const float* src = x + (((size_t)(bm0 + m) * NODES + node) * KDIM + (k0 + kk));
      v4f v = *(const v4f*)src;              // global_load_b128
      v4h h, l;
      #pragma unroll
      for (int j = 0; j < 4; ++j) {
        _Float16 hj, lj;
        split2(v[j], hj, lj);
        h[j] = hj;
        l[j] = lj;
      }
      *(v4h*)&Ahi[buf][m][kk] = h;           // ds_store_b64
      *(v4h*)&Alo[buf][m][kk] = l;
    }
    // ---- stage B (W[node]): rows k0..k0+31, all 256 cols -> store transposed [n][k] ----
    #pragma unroll
    for (int p = 0; p < 8; ++p) {
      const int kr = p * 4 + (t >> 6);       // 0..31
      const int nn = (t & 63) * 4;           // 0..252
      const float* src = W + (((size_t)node * KDIM + (k0 + kr)) * ODIM + nn);
      v4f v = *(const v4f*)src;              // global_load_b128 (coalesced over n, L2-hot)
      #pragma unroll
      for (int j = 0; j < 4; ++j) {
        _Float16 h, l;
        split2(v[j], h, l);
        Bhi[buf][nn + j][kr] = h;            // transposed scatter (merged ds stores)
        Blo[buf][nn + j][kr] = l;
      }
    }
  };

  stage(0, 0);

  const int KSTEPS = KDIM / BK;  // 8
  for (int kt = 0; kt < KSTEPS; ++kt) {
    __syncthreads();
    if (kt + 1 < KSTEPS) stage(kt + 1, (kt + 1) & 1);   // fill other buffer
    const int buf = kt & 1;

    // A frags: 16-bit A 16x32 layout -> lane(m=l&15, khalf=l>>4): two 8-elem K runs
    v16h ah[2], al[2];
    #pragma unroll
    for (int mi = 0; mi < 2; ++mi) {
      const _Float16* rh = &Ahi[buf][wm + mi * 16 + ml][kh * 8];
      const _Float16* rl = &Alo[buf][wm + mi * 16 + ml][kh * 8];
      ah[mi] = cat8(*(const v8h*)rh, *(const v8h*)(rh + 16));
      al[mi] = cat8(*(const v8h*)rl, *(const v8h*)(rl + 16));
    }

    #pragma unroll
    for (int ni = 0; ni < 8; ++ni) {
      // B frag: lane(n=l&15, khalf=l>>4): two 8-elem K runs at [n][kh*8] and [n][16+kh*8]
      const _Float16* bh = &Bhi[buf][wn + ni * 16 + ml][kh * 8];
      const _Float16* bl = &Blo[buf][wn + ni * 16 + ml][kh * 8];
      v16h bhi = cat8(*(const v8h*)bh, *(const v8h*)(bh + 16));
      v16h blo = cat8(*(const v8h*)bl, *(const v8h*)(bl + 16));
      #pragma unroll
      for (int mi = 0; mi < 2; ++mi) {
        acc[mi][ni] = __builtin_amdgcn_wmma_f32_16x16x32_f16(
            false, ah[mi], false, bhi, (short)0, acc[mi][ni], false, false);
        acc[mi][ni] = __builtin_amdgcn_wmma_f32_16x16x32_f16(
            false, ah[mi], false, blo, (short)0, acc[mi][ni], false, false);
        acc[mi][ni] = __builtin_amdgcn_wmma_f32_16x16x32_f16(
            false, al[mi], false, bhi, (short)0, acc[mi][ni], false, false);
      }
    }
  }

  // ---- epilogue: C/D 16x16 f32 layout: VGPR r -> M = kh*8 + r, N = ml ----
  const float scale = 0.0625f;  // 1/sqrt(256)
  #pragma unroll
  for (int mi = 0; mi < 2; ++mi) {
    #pragma unroll
    for (int ni = 0; ni < 8; ++ni) {
      const int ng = wn + ni * 16 + ml;
      const int mb = bm0 + wm + mi * 16 + kh * 8;
      #pragma unroll
      for (int r = 0; r < 8; ++r) {
        const size_t row = (size_t)(mb + r) * NODES + node;
        out[row * ODIM + ng] = acc[mi][ni][r] * scale;
      }
    }
  }
}

extern "C" void kernel_launch(void* const* d_in, const int* in_sizes, int n_in,
                              void* d_out, int out_size, void* d_ws, size_t ws_size,
                              hipStream_t stream) {
  (void)in_sizes; (void)n_in; (void)out_size; (void)d_ws; (void)ws_size;
  const float* x = (const float*)d_in[0];
  // d_in[1] = batch indices (int64) — implied by row layout, unused
  const float* W = (const float*)d_in[2];
  float* out = (float*)d_out;

  dim3 grid(BATCH / BM, NODES, 1);   // (32, 64) = 2048 blocks, 256 threads each
  node_linear_f16x3<<<grid, dim3(256, 1, 1), 0, stream>>>(x, W, out);
}
